// BIOULDecoder_41970420417656
// MI455X (gfx1250) — matrix operand order repaired and checked by reading, not compiled, not run on hardware
//
#include <hip/hip_runtime.h>
#include <hip/hip_bf16.h>
#include <hip/hip_fp16.h>

#define K_STATES   41
#define SEQ_LEN    1024
#define BATCH_N    1024
#define IMPOSSIBLE_V (-10000.0f)
#define P_STRIDE   66   // halves (33 dwords); 33*r mod 64 distinct for r<16 -> conflict-free frag loads
#define LOG2E      1.4426950408889634f
#define LN2        0.6931471805599453f

typedef __attribute__((ext_vector_type(16))) _Float16 v16h;
typedef __attribute__((ext_vector_type(8)))  float    v8f;

// ---- BIOUL constraint predicates (True = allowed) ----
// state 0 = O; state s>=1: q=s-1, label=q>>2, type=q&3 with 0=B,1=I,2=L,3=U
__device__ __forceinline__ bool t_allowed(int i, int j) {
  if (i == 0) {
    if (j == 0) return true;                 // O->O
    int tj = (j - 1) & 3;
    return tj == 0 || tj == 3;               // O->B, O->U
  }
  int qi = i - 1, li = qi >> 2, ti = qi & 3;
  if (j == 0) return ti == 2 || ti == 3;     // L->O, U->O
  int qj = j - 1, lj = qj >> 2, tj = qj & 3;
  switch (ti) {
    case 0: return (tj == 1 || tj == 2) && (li == lj);  // B->I, B->L (same label)
    case 1: return (tj == 1 || tj == 2) && (li == lj);  // I->I, I->L (same label)
    case 2: return tj == 0 || tj == 3;                   // L->B, L->U (any label)
    case 3: return tj == 0;                              // U->B (any label)
  }
  return false;
}
__device__ __forceinline__ bool s_allowed(int j) {       // start: forbid I, L
  if (j == 0) return true;
  int tj = (j - 1) & 3;
  return !(tj == 1 || tj == 2);
}
__device__ __forceinline__ bool e_allowed(int j) {       // end: forbid I, B
  if (j == 0) return true;
  int tj = (j - 1) & 3;
  return !(tj == 1 || tj == 0);
}

// ---- VALU cross-lane via DPP16 (row = 16 lanes = one C/D-fragment half) ----
template <int CTRL>
__device__ __forceinline__ float dpp_perm(float v) {
  int x = __builtin_amdgcn_update_dpp(0, __float_as_int(v), CTRL, 0xf, 0xf, true);
  return __int_as_float(x);
}
// full 16-lane all-reduce: quad_perm xor1, quad_perm xor2, half_mirror, mirror
__device__ __forceinline__ float rowmax16(float v) {
  v = fmaxf(v, dpp_perm<0xB1>(v));    // quad_perm:[1,0,3,2]
  v = fmaxf(v, dpp_perm<0x4E>(v));    // quad_perm:[2,3,0,1]
  v = fmaxf(v, dpp_perm<0x141>(v));   // row_half_mirror
  v = fmaxf(v, dpp_perm<0x140>(v));   // row_mirror
  return v;
}
__device__ __forceinline__ float rowsum16(float v) {
  v += dpp_perm<0xB1>(v);
  v += dpp_perm<0x4E>(v);
  v += dpp_perm<0x141>(v);
  v += dpp_perm<0x140>(v);
  return v;
}
// cross-half (xor16) needs the LDS pipe: ds_swizzle group-of-32 mode
template <int MASK>
__device__ __forceinline__ int swz_xor_i(int v) {
  return __builtin_amdgcn_ds_swizzle(v, (MASK << 10) | 0x1f);
}

// ============================================================================
// Kernel 1: CRF forward scan. 64 blocks x 1 wave32; each wave owns 16 batch
// rows. alpha held in registers in WMMA C/D layout (float a[3][8]); per step:
//   rowmax (DPP16 butterflies, VALU only) -> P = exp2(log2e*(a-m)) staged f16
//   in LDS -> A-fragment gather -> 6x v_wmma_f32_16x16x32_f16 -> branchless
//   update. E = exp(masked transitions) lives in VGPRs as B-fragments.
// Single wave per workgroup: LDS requests are processed in instruction order,
// so no barriers are needed between P stores and A-fragment loads.
// ============================================================================
__global__ __launch_bounds__(32)
void crf_scan_kernel(const float* __restrict__ emis,
                     const unsigned char* __restrict__ mask,   // jnp bool_: 1 byte/elem
                     const float* __restrict__ trans,
                     const float* __restrict__ startv,
                     const float* __restrict__ endv,
                     float* __restrict__ z_out)
{
  __shared__ __align__(16) _Float16 Pl[16 * P_STRIDE];
  const int lane = threadIdx.x;
  const int b0   = blockIdx.x * 16;
  const int hb   = lane >> 4;       // half-wave id (C/D rows 0-7 vs 8-15)
  const int lcol = lane & 15;
  unsigned int* Pu = (unsigned int*)Pl;
  char* Plb = (char*)Pl;
  const char* emis_b = (const char*)emis;

  // ---- Constant B fragments: E(64x48) = exp(masked T), zero-padded.
  // B 32x16 layout used here (contiguous k, not the ISA table order): VGPR v, lanes 0-15: K=2v,2v+1; lanes 16-31: K=16+2v,17+2v.
  union bfrag_t { _Float16 h[16]; v16h v; };
  bfrag_t bf[3][2];
  {
    const int kgrp = hb * 16;
#pragma unroll
    for (int n = 0; n < 3; ++n) {
      const int j = n * 16 + lcol;
#pragma unroll
      for (int p = 0; p < 2; ++p) {
#pragma unroll
        for (int v = 0; v < 8; ++v) {
#pragma unroll
          for (int h = 0; h < 2; ++h) {
            const int i = p * 32 + kgrp + 2 * v + h;
            float val = 0.0f;
            if (i < K_STATES && j < K_STATES && t_allowed(i, j))
              val = __expf(trans[i * K_STATES + j]);
            bf[n][p].h[2 * v + h] = (_Float16)val;
          }
        }
      }
    }
  }

  // ---- Zero P once (K-padding columns stay 0; invalid stores go to col 64).
  for (int i = lane; i < 16 * P_STRIDE / 2; i += 32) Pu[i] = 0u;

  // ---- Per-lane constants
  int   coloff[3];       // clamped byte offset of column within an emission row
  int   pbase[3];        // LDS byte address base for P stores
  float startm[3], endm[3];
  bool  cok[3];
#pragma unroll
  for (int n = 0; n < 3; ++n) {
    const int col = n * 16 + lcol;
    cok[n] = (col < K_STATES);
    coloff[n] = (cok[n] ? col : (K_STATES - 1)) * 4;   // safe dummy for invalid
    const int scol = cok[n] ? col : 64;                // invalid -> unread pad col
    pbase[n] = (hb * 8 * P_STRIDE + scol) * 2;
    startm[n] = cok[n] ? (s_allowed(col) ? startv[col] : IMPOSSIBLE_V) : 0.0f;
    endm[n]   = cok[n] ? (e_allowed(col) ? endv[col]   : IMPOSSIBLE_V) : 0.0f;
  }
  int rowoff[8];   // byte offset of (batch row, t=0) emission row; < 2^31
#pragma unroll
  for (int r = 0; r < 8; ++r)
    rowoff[r] = ((b0 + hb * 8 + r) * SEQ_LEN) * (K_STATES * 4);

  // ---- Sequence lengths (mask is a prefix mask: mask[t] = t < len)
  int len = 0;
  if (lane < 16) {
    const unsigned int* mrow = (const unsigned int*)(mask + (size_t)(b0 + lane) * SEQ_LEN);
    for (int i = 0; i < SEQ_LEN / 4; ++i) {
      unsigned int w = mrow[i];
      len += (w & 0xFF) + ((w >> 8) & 0xFF) + ((w >> 16) & 0xFF) + ((w >> 24) & 0xFF);
    }
  }
  int len8[8];
#pragma unroll
  for (int r = 0; r < 8; ++r) len8[r] = __shfl(len, hb * 8 + r);
  int lmax = len;
  lmax = max(lmax, swz_xor_i<16>(lmax));   // cross-half (one-time)
  lmax = max(lmax, swz_xor_i<8>(lmax));
  lmax = max(lmax, swz_xor_i<4>(lmax));
  lmax = max(lmax, swz_xor_i<2>(lmax));
  lmax = max(lmax, swz_xor_i<1>(lmax));
  const int tmax = __builtin_amdgcn_readfirstlane(lmax);

  // ---- alpha0 = masked_start + emissions[:,0,:]  (registers, C/D layout)
  float a[3][8];
#pragma unroll
  for (int n = 0; n < 3; ++n)
#pragma unroll
    for (int r = 0; r < 8; ++r)
      a[n][r] = startm[n] + *(const float*)(emis_b + (rowoff[r] + coloff[n]));

  const int adw = hb * 4;   // A-fragment dword group (x2 = dword offset 8*hb: contiguous k, same as B)
  // incremental prefetch offset: row (b0+lane), time t+1 (valid for lane<16)
  int pfoff = ((b0 + lane) * SEQ_LEN + 2) * (K_STATES * 4);

  for (int t = 1; t < tmax; ++t) {
    // prefetch next step's emission rows (global_prefetch_b8)
    if (lane < 16 && (t + 1) < SEQ_LEN) {
      __builtin_prefetch(emis_b + pfoff, 0, 1);
      __builtin_prefetch(emis_b + pfoff + 128, 0, 1);
    }
    pfoff += K_STATES * 4;

    // ---- rowmax via DPP16 butterflies (VALU only; rows match DPP rows)
    float m8[8];
#pragma unroll
    for (int r = 0; r < 8; ++r) {
      float v = fmaxf(a[0][r], a[1][r]);
      v = fmaxf(v, cok[2] ? a[2][r] : -3.0e38f);
      m8[r] = rowmax16(v);
    }

    // ---- P = exp2(log2e*(alpha - m)) staged as f16 (branchless; invalid->pad)
#pragma unroll
    for (int n = 0; n < 3; ++n) {
#pragma unroll
      for (int r = 0; r < 8; ++r) {
        float p = __builtin_amdgcn_exp2f(LOG2E * (a[n][r] - m8[r]));
        if (!cok[n]) p = 0.0f;
        *(_Float16*)(Plb + pbase[n] + r * (P_STRIDE * 2)) = (_Float16)p;
      }
    }

    // ---- gather A fragments (16x32 f16 x2) in contiguous k order (k = 16*hb + s), matching B
    // (same-wave LDS ordering guarantees these loads see the stores above)
    union afrag_t { unsigned int u[8]; v16h v; };
    afrag_t alo, ahi;
#pragma unroll
    for (int v = 0; v < 8; ++v) {
      const int d = adw * 2 + v;
      alo.u[v] = Pu[lcol * 33 + d];
      ahi.u[v] = Pu[lcol * 33 + d + 16];
    }

    // ---- GEMM tiles + branchless alpha update
    const int toff = t * (K_STATES * 4);
#pragma unroll
    for (int n = 0; n < 3; ++n) {
      v8f acc = {};
      acc = __builtin_amdgcn_wmma_f32_16x16x32_f16(false, alo.v, false, bf[n][0].v,
                                                   (short)0, acc, false, false);
      acc = __builtin_amdgcn_wmma_f32_16x16x32_f16(false, ahi.v, false, bf[n][1].v,
                                                   (short)0, acc, false, false);
#pragma unroll
      for (int r = 0; r < 8; ++r) {
        const float em = *(const float*)(emis_b + (rowoff[r] + toff + coloff[n]));
        const float l2 = __builtin_amdgcn_logf(acc[r] + 1.0e-30f);  // log(0) guard
        const float nv = fmaf(LN2, l2, m8[r] + em);
        a[n][r] = (t < len8[r]) ? nv : a[n][r];
      }
    }
  }

  // ---- z = logsumexp(alpha_T + masked_end) per batch row (DPP reductions)
#pragma unroll
  for (int r = 0; r < 8; ++r) {
    const float v0 = a[0][r] + endm[0];
    const float v1 = a[1][r] + endm[1];
    const float v2 = cok[2] ? (a[2][r] + endm[2]) : -3.0e38f;
    const float mx = rowmax16(fmaxf(fmaxf(v0, v1), v2));
    float s = __builtin_amdgcn_exp2f(LOG2E * (v0 - mx))
            + __builtin_amdgcn_exp2f(LOG2E * (v1 - mx))
            + (cok[2] ? __builtin_amdgcn_exp2f(LOG2E * (v2 - mx)) : 0.0f);
    s = rowsum16(s);
    if (lcol == 0)
      z_out[b0 + hb * 8 + r] = fmaf(LN2, __builtin_amdgcn_logf(s), mx);
  }
}

// ============================================================================
// Kernel 2: gold-path score per batch, nll[b] = score[b] - z[b]
// ============================================================================
__global__ __launch_bounds__(256)
void crf_score_kernel(const float* __restrict__ emis,
                      const unsigned char* __restrict__ mask,
                      const int* __restrict__ tags,
                      const float* __restrict__ trans,
                      const float* __restrict__ startv,
                      const float* __restrict__ endv,
                      const float* __restrict__ z,
                      float* __restrict__ nll)
{
  const int b = blockIdx.x * blockDim.x + threadIdx.x;
  if (b >= BATCH_N) return;
  const size_t base = (size_t)b * SEQ_LEN;
  int t0 = tags[base];
  float sc = (s_allowed(t0) ? startv[t0] : IMPOSSIBLE_V)
           + emis[base * K_STATES + t0];
  int last = t0;
  for (int t = 1; t < SEQ_LEN; ++t) {
    if (!mask[base + t]) break;                 // prefix mask
    const int tg = tags[base + t];
    const int pv = tags[base + t - 1];
    const float tr = t_allowed(pv, tg) ? trans[pv * K_STATES + tg] : IMPOSSIBLE_V;
    sc += tr + emis[(base + t) * K_STATES + tg];
    last = tg;
  }
  sc += e_allowed(last) ? endv[last] : IMPOSSIBLE_V;
  nll[b] = sc - z[b];
}

// ============================================================================
// Kernel 3: mean over 1024 -> d_out[0]
// ============================================================================
__global__ __launch_bounds__(256)
void crf_reduce_kernel(const float* __restrict__ nll, float* __restrict__ out)
{
  __shared__ float s[256];
  const int tid = threadIdx.x;
  float v = nll[tid] + nll[tid + 256] + nll[tid + 512] + nll[tid + 768];
  s[tid] = v;
  __syncthreads();
#pragma unroll
  for (int o = 128; o > 0; o >>= 1) {
    if (tid < o) s[tid] += s[tid + o];
    __syncthreads();
  }
  if (tid == 0) out[0] = s[0] / (float)BATCH_N;
}

extern "C" void kernel_launch(void* const* d_in, const int* in_sizes, int n_in,
                              void* d_out, int out_size, void* d_ws, size_t ws_size,
                              hipStream_t stream)
{
  const float*         emis  = (const float*)d_in[0];          // (1024,1024,41) f32
  const unsigned char* mask  = (const unsigned char*)d_in[1];  // (1024,1024) bool (1B)
  const int*           tags  = (const int*)d_in[2];            // (1024,1024) i32
  const float*         trans = (const float*)d_in[3];          // (41,41) f32
  const float*         startv= (const float*)d_in[4];          // (41,) f32
  const float*         endv  = (const float*)d_in[5];          // (41,) f32
  float* out = (float*)d_out;

  float* z_ws   = (float*)d_ws;             // 1024 f32
  float* nll_ws = z_ws + BATCH_N;           // 1024 f32

  crf_scan_kernel<<<BATCH_N / 16, 32, 0, stream>>>(emis, mask, trans, startv, endv, z_ws);
  crf_score_kernel<<<(BATCH_N + 255) / 256, 256, 0, stream>>>(emis, mask, tags, trans,
                                                              startv, endv, z_ws, nll_ws);
  crf_reduce_kernel<<<1, 256, 0, stream>>>(nll_ws, out);
}
